// InvariantPointAttention_75161927680196
// MI455X (gfx1250) — matrix operand order repaired and checked by reading, not compiled, not run on hardware
//
#include <hip/hip_runtime.h>
#include <hip/hip_bf16.h>
#include <stdint.h>

typedef __attribute__((ext_vector_type(16))) __bf16 v16bf;
typedef __attribute__((ext_vector_type(8)))  float  v8f;

#define Nn 1024
#define Dd 384
#define Hh 8
#define ROWP 1028   // padded LDS row stride (floats): 4-float pad => conflict-free column gathers

// ---------------------------------------------------------------------------
// K1: projections + rigid transform -> Qe/Ke (bf16, 32-dim fused q/k vectors),
//     sv (bf16 16), pv_global (bf16 padded 16), per-row biases.
// ---------------------------------------------------------------------------
__global__ void k1_proj(const float* __restrict__ x, const float* __restrict__ rot,
                        const float* __restrict__ trans,
                        const float* __restrict__ Wsq, const float* __restrict__ Wsk,
                        const float* __restrict__ Wsv, const float* __restrict__ Wpq,
                        const float* __restrict__ Wpk, const float* __restrict__ Wpv,
                        const float* __restrict__ pwv,
                        __bf16* __restrict__ Qe, __bf16* __restrict__ Ke,
                        __bf16* __restrict__ Svb, __bf16* __restrict__ Pvb,
                        float* __restrict__ biasq, float* __restrict__ biask)
{
    int i = blockIdx.x;
    __shared__ float xs[Dd];
    __shared__ float sq[128], sk[128], sv[128];
    __shared__ float pq[96], pk[96], pv[96];
    __shared__ float pqg[96], pkg[96], pvg[96];
    int t = threadIdx.x, nt = blockDim.x;

    for (int d = t; d < Dd; d += nt) xs[d] = x[i * Dd + d];
    __syncthreads();

    for (int col = t; col < 672; col += nt) {
        const float* W; float* dst; int c; int ncols;
        if      (col < 128) { W = Wsq; c = col;       dst = sq; ncols = 128; }
        else if (col < 256) { W = Wsk; c = col - 128; dst = sk; ncols = 128; }
        else if (col < 384) { W = Wsv; c = col - 256; dst = sv; ncols = 128; }
        else if (col < 480) { W = Wpq; c = col - 384; dst = pq; ncols = 96;  }
        else if (col < 576) { W = Wpk; c = col - 480; dst = pk; ncols = 96;  }
        else                { W = Wpv; c = col - 576; dst = pv; ncols = 96;  }
        float acc = 0.f;
        for (int d = 0; d < Dd; ++d) acc += xs[d] * W[d * ncols + c];
        dst[c] = acc;
    }
    __syncthreads();

    // global-frame points: out_c = sum_j rot[i][c][j]*p[j] + trans[i][c]
    for (int zz = t; zz < 288; zz += nt) {
        int arr = zz / 96, z = zz % 96, c = z % 3, base = z - c;
        const float* src = (arr == 0) ? pq : (arr == 1) ? pk : pv;
        float* dst       = (arr == 0) ? pqg : (arr == 1) ? pkg : pvg;
        float a = trans[i * 3 + c];
        for (int j = 0; j < 3; ++j) a += rot[i * 9 + c * 3 + j] * src[base + j];
        dst[z] = a;
    }
    __syncthreads();

    for (int z = t; z < Hh * 32; z += nt) {           // Qe / Ke
        int h = z >> 5, e = z & 31;
        float pw = log1pf(expf(pwv[h]));
        float v, w;
        if      (e < 16) { v = sq[h * 16 + e] * 0.25f;        w = sk[h * 16 + e]; }
        else if (e < 28) { v = pw * pqg[h * 12 + (e - 16)];   w = pkg[h * 12 + (e - 16)]; }
        else             { v = 0.f;                           w = 0.f; }
        Qe[((size_t)h * Nn + i) * 32 + e] = (__bf16)v;
        Ke[((size_t)h * Nn + i) * 32 + e] = (__bf16)w;
    }
    for (int z = t; z < Hh * 16; z += nt) {           // sv / pv_global
        int h = z >> 4, e = z & 15;
        Svb[((size_t)h * Nn + i) * 16 + e] = (__bf16)sv[h * 16 + e];
        Pvb[((size_t)h * Nn + i) * 16 + e] = (__bf16)((e < 12) ? pvg[h * 12 + e] : 0.f);
    }
    for (int h = t; h < Hh; h += nt) {                // biases -0.5*pw*|q|^2, |k|^2
        float pw = log1pf(expf(pwv[h]));
        float q2 = 0.f, k2 = 0.f;
        for (int z = 0; z < 12; ++z) {
            q2 += pqg[h * 12 + z] * pqg[h * 12 + z];
            k2 += pkg[h * 12 + z] * pkg[h * 12 + z];
        }
        biasq[h * Nn + i] = -0.5f * pw * q2;
        biask[h * Nn + i] = -0.5f * pw * k2;
    }
}

// ---------------------------------------------------------------------------
// K2: pair bias via WMMA:  S[h][i][j] = sum_c pair[i][j][c] * W_pair[c][h]
// one wave per (i, 64-wide j group); B fragments (W_pair) loaded once.
// ---------------------------------------------------------------------------
__global__ void k2_pairbias(const float* __restrict__ pair, const float* __restrict__ Wpair,
                            float* __restrict__ S)
{
    int i    = blockIdx.x;
    int jg   = blockIdx.y;            // 16 groups of 4 j-tiles
    int lane = threadIdx.x & 31;
    int mlo  = lane & 15, half = lane >> 4;
    int n    = mlo;

    v16bf b0, b1;
#pragma unroll
    for (int e = 0; e < 16; ++e) {
        int k = (half << 4) + e;
        b0[e] = (__bf16)((n < 8) ? Wpair[k * Hh + n]        : 0.f);
        b1[e] = (__bf16)((n < 8) ? Wpair[(32 + k) * Hh + n] : 0.f);
    }

    for (int jbi = 0; jbi < 4; ++jbi) {
        int jb = jg * 4 + jbi;
        v16bf a0, a1;
        const float* prow = pair + ((size_t)i * Nn + jb * 16 + mlo) * 64;
#pragma unroll
        for (int e = 0; e < 16; ++e) {
            int k = (half << 4) + e;
            a0[e] = (__bf16)prow[k];
            a1[e] = (__bf16)prow[32 + k];
        }
        v8f acc = {};
        acc = __builtin_amdgcn_wmma_f32_16x16x32_bf16(false, a0, false, b0, (short)0, acc, false, false);
        acc = __builtin_amdgcn_wmma_f32_16x16x32_bf16(false, a1, false, b1, (short)0, acc, false, false);
        if (n < 8) {
#pragma unroll
            for (int r = 0; r < 8; ++r) {
                int j = jb * 16 + r + 8 * half;
                S[((size_t)n * Nn + i) * Nn + j] = acc[r];
            }
        }
    }
}

// ---------------------------------------------------------------------------
// K3 (fused): logits (WMMA) -> LDS strip -> softmax (wave32 shuffles)
//             -> attn written once to S -> attn@sv, attn@pv (WMMA from LDS)
// one wave per (i-block, head); 16x1024 f32 strip in dynamic LDS.
// ---------------------------------------------------------------------------
__global__ void k3_fused(const __bf16* __restrict__ Qe, const __bf16* __restrict__ Ke,
                         const float* __restrict__ biasq, const float* __restrict__ biask,
                         const unsigned char* __restrict__ mask,
                         const __bf16* __restrict__ Svb, const __bf16* __restrict__ Pvb,
                         float* __restrict__ S, float* __restrict__ feat,
                         float* __restrict__ pva)
{
    extern __shared__ float att[];   // [16][ROWP]
    int ib = blockIdx.x, h = blockIdx.y;
    int lane = threadIdx.x & 31;
    int mlo = lane & 15, half = lane >> 4;
    int n = mlo;

    // ---- phase 1: logits tile-by-tile into LDS ----
    v16bf a;
    const __bf16* qrow = Qe + ((size_t)h * Nn + ib * 16 + mlo) * 32;
#pragma unroll
    for (int e = 0; e < 16; ++e) { int k = (half << 4) + e; a[e] = qrow[k]; }

    float bq[8];
#pragma unroll
    for (int r = 0; r < 8; ++r) bq[r] = biasq[h * Nn + ib * 16 + r + 8 * half];

    for (int jb = 0; jb < Nn / 16; ++jb) {
        v16bf b;
        const __bf16* krow = Ke + ((size_t)h * Nn + jb * 16 + n) * 32;
#pragma unroll
        for (int e = 0; e < 16; ++e) { int k = (half << 4) + e; b[e] = krow[k]; }
        v8f acc = {};
        acc = __builtin_amdgcn_wmma_f32_16x16x32_bf16(false, a, false, b, (short)0, acc, false, false);
        int j = jb * 16 + n;
        float bk = biask[h * Nn + j];
        bool mk = mask[j] != 0;
#pragma unroll
        for (int r = 0; r < 8; ++r) {
            int ml = r + 8 * half;     // local row 0..15
            float pb = S[((size_t)h * Nn + ib * 16 + ml) * Nn + j];   // pair bias
            float v = acc[r] + bq[r] + bk + pb;
            att[ml * ROWP + j] = mk ? v : -1.0e9f;
        }
    }
    __syncthreads();

    // ---- phase 2: softmax per row; write attn once to S (for k7) ----
    for (int m = 0; m < 16; ++m) {
        float mx = -3.4e38f;
        for (int j = lane; j < Nn; j += 32) mx = fmaxf(mx, att[m * ROWP + j]);
#pragma unroll
        for (int s = 16; s > 0; s >>= 1) mx = fmaxf(mx, __shfl_xor(mx, s, 32));
        float sm = 0.f;
        for (int j = lane; j < Nn; j += 32) {
            float e = __expf(att[m * ROWP + j] - mx);
            att[m * ROWP + j] = e;
            sm += e;
        }
#pragma unroll
        for (int s = 16; s > 0; s >>= 1) sm += __shfl_xor(sm, s, 32);
        float inv = 1.0f / sm;
        float* Srow = S + ((size_t)h * Nn + ib * 16 + m) * Nn;
        for (int j = lane; j < Nn; j += 32) {
            float v = att[m * ROWP + j] * inv;
            att[m * ROWP + j] = v;
            Srow[j] = v;
        }
    }
    __syncthreads();

    // ---- phase 3: so = attn @ sv, pva = attn @ pv_g, A from LDS ----
    v8f aso = {}, apv = {};
    for (int j0 = 0; j0 < Nn; j0 += 32) {
        v16bf aa, bs, bp;
#pragma unroll
        for (int e = 0; e < 16; ++e) {
            int k = (half << 4) + e;
            aa[e] = (__bf16)att[mlo * ROWP + j0 + k];
        }
#pragma unroll
        for (int e = 0; e < 16; ++e) {
            int j = j0 + (half << 4) + e;
            bs[e] = Svb[((size_t)h * Nn + j) * 16 + mlo];
            bp[e] = Pvb[((size_t)h * Nn + j) * 16 + mlo];
        }
        aso = __builtin_amdgcn_wmma_f32_16x16x32_bf16(false, aa, false, bs, (short)0, aso, false, false);
        apv = __builtin_amdgcn_wmma_f32_16x16x32_bf16(false, aa, false, bp, (short)0, apv, false, false);
    }
#pragma unroll
    for (int r = 0; r < 8; ++r) {
        int irow = ib * 16 + r + 8 * half;
        feat[(size_t)irow * 768 + h * 16 + n] = aso[r];
        pva[((size_t)h * Nn + irow) * 16 + n] = apv[r];
    }
}

// ---------------------------------------------------------------------------
// K6: local-frame points pvl = R^T (pva - t), norms
// ---------------------------------------------------------------------------
__global__ void k6_pvl(const float* __restrict__ pva, const float* __restrict__ rot,
                       const float* __restrict__ trans, float* __restrict__ feat)
{
    int i = blockIdx.x, t = threadIdx.x;   // 96 threads: h*12 + p*3 + c
    int h = t / 12, z = t % 12, p = z / 3, c = z % 3;
    float v = 0.f;
    for (int j = 0; j < 3; ++j) {
        float d = pva[((size_t)h * Nn + i) * 16 + p * 3 + j] - trans[i * 3 + j];
        v += rot[i * 9 + j * 3 + c] * d;
    }
    feat[(size_t)i * 768 + 128 + h * 12 + p * 3 + c] = v;
    if (c == 0) {
        float s = 0.f;
        for (int cc = 0; cc < 3; ++cc) {
            float vv = 0.f;
            for (int j = 0; j < 3; ++j) {
                float d = pva[((size_t)h * Nn + i) * 16 + p * 3 + j] - trans[i * 3 + j];
                vv += rot[i * 9 + j * 3 + cc] * d;
            }
            s += vv * vv;
        }
        feat[(size_t)i * 768 + 224 + h * 4 + p] = sqrtf(s);
    }
}

// ---------------------------------------------------------------------------
// K7: po[h][i][c] = sum_j attn[h][i][j] * pair[i][j][c]  (bandwidth pass)
// ---------------------------------------------------------------------------
__global__ void k7_po(const float* __restrict__ S, const float* __restrict__ pair,
                      float* __restrict__ feat)
{
    int i = blockIdx.x, t = threadIdx.x;   // 256 threads, 2 outputs each
    int h0 = t >> 6, c = t & 63, h1 = h0 + 4;
    const float* prow = pair + (size_t)i * Nn * 64;
    const float* a0 = S + ((size_t)h0 * Nn + i) * Nn;
    const float* a1 = S + ((size_t)h1 * Nn + i) * Nn;
    float acc0 = 0.f, acc1 = 0.f;
    for (int j = 0; j < Nn; ++j) {
        float pv = prow[(size_t)j * 64 + c];
        acc0 += a0[j] * pv;
        acc1 += a1[j] * pv;
    }
    feat[(size_t)i * 768 + 256 + h0 * 64 + c] = acc0;
    feat[(size_t)i * 768 + 256 + h1 * 64 + c] = acc1;
}

// ---------------------------------------------------------------------------
// K8: out = feat(1024x768) @ W_out(768x384) + b_out   (WMMA)
// ---------------------------------------------------------------------------
__global__ void k8_out(const float* __restrict__ feat, const float* __restrict__ Wout,
                       const float* __restrict__ bout, float* __restrict__ out)
{
    int ib = blockIdx.x, ob = blockIdx.y;
    int lane = threadIdx.x & 31;
    int mlo = lane & 15, half = lane >> 4;

    v8f acc = {};
    for (int k0 = 0; k0 < 768; k0 += 32) {
        v16bf a, b;
        const float* frow = feat + ((size_t)ib * 16 + mlo) * 768 + k0;
#pragma unroll
        for (int e = 0; e < 16; ++e) {
            int k = (half << 4) + e;
            a[e] = (__bf16)frow[k];
        }
#pragma unroll
        for (int e = 0; e < 16; ++e) {
            int k = k0 + (half << 4) + e;
            b[e] = (__bf16)Wout[(size_t)k * Dd + ob * 16 + mlo];
        }
        acc = __builtin_amdgcn_wmma_f32_16x16x32_bf16(false, a, false, b, (short)0, acc, false, false);
    }
    int n = mlo;
    float bo = bout[ob * 16 + n];
#pragma unroll
    for (int r = 0; r < 8; ++r) {
        int irow = ib * 16 + r + 8 * half;
        out[(size_t)irow * Dd + ob * 16 + n] = acc[r] + bo;
    }
}

// ---------------------------------------------------------------------------
extern "C" void kernel_launch(void* const* d_in, const int* in_sizes, int n_in,
                              void* d_out, int out_size, void* d_ws, size_t ws_size,
                              hipStream_t stream)
{
    const float* x     = (const float*)d_in[0];
    const float* pair  = (const float*)d_in[1];
    const float* rot   = (const float*)d_in[2];
    const float* trans = (const float*)d_in[3];
    const unsigned char* mask = (const unsigned char*)d_in[4];
    const float* Wsq   = (const float*)d_in[5];
    const float* Wsk   = (const float*)d_in[6];
    const float* Wsv   = (const float*)d_in[7];
    const float* Wpq   = (const float*)d_in[8];
    const float* Wpk   = (const float*)d_in[9];
    const float* Wpv   = (const float*)d_in[10];
    const float* Wpair = (const float*)d_in[11];
    const float* pwv   = (const float*)d_in[12];
    const float* Wout  = (const float*)d_in[13];
    const float* bout  = (const float*)d_in[14];
    float* out = (float*)d_out;

    char* p = (char*)d_ws;
    __bf16* Qe   = (__bf16*)p;  p += (size_t)Hh * Nn * 32 * 2;
    __bf16* Ke   = (__bf16*)p;  p += (size_t)Hh * Nn * 32 * 2;
    __bf16* Svb  = (__bf16*)p;  p += (size_t)Hh * Nn * 16 * 2;
    __bf16* Pvb  = (__bf16*)p;  p += (size_t)Hh * Nn * 16 * 2;
    float* biasq = (float*)p;   p += (size_t)Hh * Nn * 4;
    float* biask = (float*)p;   p += (size_t)Hh * Nn * 4;
    float* S     = (float*)p;   p += (size_t)Hh * Nn * Nn * 4;   // pairbias -> attn (in place)
    float* pva   = (float*)p;   p += (size_t)Hh * Nn * 16 * 4;
    float* feat  = (float*)p;   p += (size_t)Nn * 768 * 4;

    size_t ldsBytes = (size_t)16 * ROWP * sizeof(float);   // 65792 B strip

    k1_proj<<<Nn, 128, 0, stream>>>(x, rot, trans, Wsq, Wsk, Wsv, Wpq, Wpk, Wpv, pwv,
                                    Qe, Ke, Svb, Pvb, biasq, biask);
    k2_pairbias<<<dim3(Nn, Nn / 64), 32, 0, stream>>>(pair, Wpair, S);
    k3_fused<<<dim3(Nn / 16, Hh), 32, ldsBytes, stream>>>(Qe, Ke, biasq, biask, mask,
                                                          Svb, Pvb, S, feat, pva);
    k6_pvl<<<Nn, 96, 0, stream>>>(pva, rot, trans, feat);
    k7_po<<<Nn, 256, 0, stream>>>(S, pair, feat);
    k8_out<<<dim3(Nn / 16, Dd / 16), 32, 0, stream>>>(feat, Wout, bout, out);
}
